// GlobalAttention_29841432773285
// MI455X (gfx1250) — matrix operand order repaired and checked by reading, not compiled, not run on hardware
//
#include <hip/hip_runtime.h>
#include <hip/hip_bf16.h>
#include <stdint.h>

#define R_   256
#define N_   2048
#define CIN  256
#define CH   64
#define NH   8
#define HC   512           // NH*CH
#define INFV 1e9f

typedef __attribute__((ext_vector_type(16))) __bf16 v16bf;
typedef __attribute__((ext_vector_type(8)))  float  v8f;

// native bf16 convert: lowers to v_cvt_pk_bf16_f32 on gfx1250
__device__ __forceinline__ __bf16 f2bf(float x) { return (__bf16)x; }

__device__ __forceinline__ v8f wmma_bf16(v16bf a, v16bf b, v8f c) {
  return __builtin_amdgcn_wmma_f32_16x16x32_bf16(false, a, false, b, (short)0, c, false, false);
}

// ---- A fragment (16x32, M x K), contiguous k order (k = 16*(lane/16) + s) ----
// lanes 0-15: M=lane, Kbase=0 ; lanes 16-31: M=lane-16, Kbase=16
// v0..v7 -> K = Kbase+0..15 (pairs)
__device__ __forceinline__ v16bf ldA_f32(const float* base, int ld, int lane) {
  int m  = lane & 15;
  int kb = (lane & 16) ? 16 : 0;
  const float* rp = base + (size_t)m * ld;
  v16bf a;
#pragma unroll
  for (int i = 0; i < 4; ++i) {
    a[2*i]     = f2bf(rp[kb + 2*i]);
    a[2*i + 1] = f2bf(rp[kb + 2*i + 1]);
    a[8 + 2*i]     = f2bf(rp[kb + 8 + 2*i]);
    a[8 + 2*i + 1] = f2bf(rp[kb + 8 + 2*i + 1]);
  }
  return a;
}

__device__ __forceinline__ v16bf ldA_bf16(const __bf16* base, int ld, int lane) {
  int m  = lane & 15;
  int kb = (lane & 16) ? 16 : 0;
  const __bf16* rp = base + (size_t)m * ld;
  v16bf a;
#pragma unroll
  for (int i = 0; i < 4; ++i) {
    a[2*i]     = rp[kb + 2*i];
    a[2*i + 1] = rp[kb + 2*i + 1];
    a[8 + 2*i]     = rp[kb + 8 + 2*i];
    a[8 + 2*i + 1] = rp[kb + 8 + 2*i + 1];
  }
  return a;
}

// ---- B fragment (32x16, K x N), contiguous k order, same as A ----
// lanes 0-15: N=lane, K=0..15 across v0..v7 (pairs); lanes 16-31: K=16..31
// source is transposed bf16 [N][K] row-major (K contiguous)
__device__ __forceinline__ v16bf ldB_T_bf16(const __bf16* base, int ldk, int lane) {
  int n    = lane & 15;
  int koff = (lane & 16) ? 16 : 0;
  const __bf16* rp = base + (size_t)n * ldk + koff;
  v16bf b;
#pragma unroll
  for (int v = 0; v < 8; ++v) {
    b[2*v]     = rp[2*v];
    b[2*v + 1] = rp[2*v + 1];
  }
  return b;
}
// source is fp32 [K][N] row-major (K-major, stride ldn)
__device__ __forceinline__ v16bf ldB_f32_kmaj(const float* base, int ldn, int lane) {
  int n    = lane & 15;
  int koff = (lane & 16) ? 16 : 0;
  const float* cp = base + n + (size_t)koff * ldn;
  v16bf b;
#pragma unroll
  for (int v = 0; v < 8; ++v) {
    b[2*v]     = f2bf(cp[(2*v)     * ldn]);
    b[2*v + 1] = f2bf(cp[(2*v + 1) * ldn]);
  }
  return b;
}

// cooperative global->LDS copy, 16B granules (nels must be multiple of 2048)
__device__ __forceinline__ void coop_copy_bf16(__bf16* dst, const __bf16* src,
                                               int nels, int tid) {
  const uint4* s = (const uint4*)src;
  uint4*       d = (uint4*)dst;
  int nvec = nels >> 3;
#pragma unroll 4
  for (int i = tid; i < nvec; i += 256) d[i] = s[i];
}

// =====================  kernel 0: weight convert/transpose  =====================
// W [K][Ncols] fp32 -> WT [Ncols][K] bf16
__global__ void wconv_kernel(const float* __restrict__ W, __bf16* __restrict__ WT,
                             int K, int Ncols) {
  int idx = blockIdx.x * 256 + threadIdx.x;
  if (idx < K * Ncols) {
    int k = idx / Ncols, j = idx % Ncols;
    WT[(size_t)j * K + k] = f2bf(W[idx]);
  }
}

// =====================  kernel 1: fully fused per-row global attention  =====================
// LDS layout (dynamic, phase-overlaid):
//  prologue:   mbar[256] f32 @0, redbuf[256] f32 @1024, q_lds[512] f32 @2048 (in logits region)
//  phase A/B:  logits [8][2048] f32   @ 0       (64KB)
//              aprob  [16][2048] bf16 @ 65536   (64KB)
//              qk     [256][16] f32   @ 131072  (16KB)
//              amst   [16][256] bf16  @ 147456  (8KB)
//  phase C:    pstage 8*[32][512] bf16 @ 0      (256KB, overlays dead A/B regions)
//              wbuf   32KB bf16        @ 262144
//  persistent: olds   [16][64] f32     @ 294912 (4KB)
#define SMEM_BYTES 299008

__global__ __launch_bounds__(256) void attn_row_kernel(
    const float* __restrict__ m, const float* __restrict__ mask,
    const float* __restrict__ Wq, const float* __restrict__ Wk,
    const __bf16* __restrict__ WgT, const __bf16* __restrict__ WoT,
    const __bf16* __restrict__ WvT,
    const float* __restrict__ bg, const float* __restrict__ bo,
    float* __restrict__ out) {
  extern __shared__ char smem[];
  float*  mbar   = (float*) (smem);
  float*  redbuf = (float*) (smem + 1024);
  float*  q_lds  = (float*) (smem + 2048);
  float*  logits = (float*) (smem);
  __bf16* aprob  = (__bf16*)(smem + 65536);
  float*  qk     = (float*) (smem + 131072);
  __bf16* amst   = (__bf16*)(smem + 147456);
  __bf16* pstage = (__bf16*)(smem);
  __bf16* wbuf   = (__bf16*)(smem + 262144);
  float*  olds   = (float*) (smem + 294912);

  const int r = blockIdx.x, tid = threadIdx.x;
  const int lane = tid & 31, wave = tid >> 5;
  const int colL = lane & 15, rb = (lane & 16) ? 8 : 0;
  const float* mr = m + (size_t)r * N_ * CIN;
  const float* maskr = mask + (size_t)r * N_;

  // ---- fused masked-mean pool + q = (mbar@Wq)*c^-1/2 (brings m[r] into L2) ----
  {
    float acc = 0.f, cnt = 0.f;
#pragma unroll 4
    for (int n = 0; n < N_; ++n)
      acc += mr[(size_t)n * CIN + tid] * maskr[n];
    for (int n = tid; n < N_; n += 256) cnt += maskr[n];
    redbuf[tid] = cnt;
    __syncthreads();
    for (int s = 128; s; s >>= 1) { if (tid < s) redbuf[tid] += redbuf[tid + s]; __syncthreads(); }
    float denom = redbuf[0] + 1e-10f;
    mbar[tid] = acc / denom;
    __syncthreads();
    const float scale = 0.125f;  // 64^-0.5
    for (int j = tid; j < HC; j += 256) {
      float a = 0.f;
      for (int C = 0; C < CIN; ++C) a += mbar[C] * Wq[(size_t)C * HC + j];
      q_lds[j] = a * scale;
    }
  }
  __syncthreads();

  // ---- qk[C][h] = sum_c Wk[C][c] * q[h][c]   (fold Wk into q) ----
  for (int h = 0; h < 16; ++h) {
    float a = 0.f;
    if (h < NH)
      for (int c = 0; c < CH; ++c) a += Wk[(size_t)tid * CH + c] * q_lds[h * CH + c];
    qk[tid * 16 + h] = a;
  }
  __syncthreads();   // mbar/q_lds dead; logits region free

  // ---- phase A: logits = m[r] @ qk + INF*(mask-1) ----
  for (int t = wave; t < N_ / 16; t += 8) {
    int n0 = t * 16;
    v8f acc = {};
#pragma unroll
    for (int kt = 0; kt < CIN / 32; ++kt) {
      v16bf A = ldA_f32(mr + (size_t)n0 * CIN + kt * 32, CIN, lane);
      v16bf Bf = ldB_f32_kmaj(qk + kt * 32 * 16, 16, lane);
      acc = wmma_bf16(A, Bf, acc);
    }
    if (colL < NH) {
#pragma unroll
      for (int i = 0; i < 8; ++i) {
        int n = n0 + rb + i;
        logits[colL * N_ + n] = acc[i] + INFV * (maskr[n] - 1.f);
      }
    }
  }
  __syncthreads();

  // ---- softmax per head (wave w <-> head w) ----
  {
    int h = wave;
    float mx = -3.0e38f;
    for (int n = lane; n < N_; n += 32) mx = fmaxf(mx, logits[h * N_ + n]);
#pragma unroll
    for (int off = 16; off; off >>= 1) mx = fmaxf(mx, __shfl_xor(mx, off, 32));
    float sum = 0.f;
    for (int n = lane; n < N_; n += 32) sum += __expf(logits[h * N_ + n] - mx);
#pragma unroll
    for (int off = 16; off; off >>= 1) sum += __shfl_xor(sum, off, 32);
    float inv = 1.f / (sum + 1e-10f);
    for (int n = lane; n < N_; n += 32)
      aprob[h * N_ + n] = f2bf(__expf(logits[h * N_ + n] - mx) * inv);
  }
  for (int idx = tid; idx < 8 * N_; idx += 256) aprob[8 * N_ + idx] = f2bf(0.f); // pad heads
  __syncthreads();

  // ---- phase B: am = aprob @ m[r]  ([16 x 2048] @ [2048 x 256]) ----
  {
    int jt0 = wave, jt1 = wave + 8;
    v8f acc0 = {}, acc1 = {};
    for (int kt = 0; kt < N_ / 32; ++kt) {
      v16bf A = ldA_bf16(aprob + kt * 32, N_, lane);
      acc0 = wmma_bf16(A, ldB_f32_kmaj(mr + (size_t)kt * 32 * CIN + jt0 * 16, CIN, lane), acc0);
      acc1 = wmma_bf16(A, ldB_f32_kmaj(mr + (size_t)kt * 32 * CIN + jt1 * 16, CIN, lane), acc1);
    }
    int c0 = jt0 * 16 + colL, c1 = jt1 * 16 + colL;
#pragma unroll
    for (int i = 0; i < 8; ++i) {
      amst[(rb + i) * CIN + c0] = f2bf(acc0[i]);
      amst[(rb + i) * CIN + c1] = f2bf(acc1[i]);
    }
  }
  __syncthreads();

  // ---- o = am @ Wv  ([16 x 256] @ [256 x 64]) -> olds[16][64] ----
  if (wave < 4) {
    int jt = wave;
    v8f acc = {};
#pragma unroll
    for (int kt = 0; kt < CIN / 32; ++kt) {
      v16bf A  = ldA_bf16(amst + kt * 32, CIN, lane);
      v16bf Bf = ldB_T_bf16(WvT + (size_t)(jt * 16) * CIN + kt * 32, CIN, lane);
      acc = wmma_bf16(A, Bf, acc);
    }
    int col = jt * 16 + colL;
#pragma unroll
    for (int i = 0; i < 8; ++i) olds[(rb + i) * CH + col] = acc[i];
  }
  __syncthreads();   // after this barrier phase A/B LDS regions are dead

  // ---- phase C: G = sigmoid(m@Wg + bg) ; P = G*o ; out = P@Wo + bo ----
  // 8 waves x 32 rows = 256 rows per superblock iteration; weight chunks staged in LDS
  __bf16* pst = pstage + (size_t)wave * 32 * HC;   // [32][512]
  for (int it = 0; it < 8; ++it) {
    int nbase = it * 256 + wave * 32;

    // cache A fragments for this wave's 2 row-tiles (m, fp32 -> bf16)
    v16bf Am[2][8];
#pragma unroll
    for (int tl = 0; tl < 2; ++tl)
#pragma unroll
      for (int kt = 0; kt < 8; ++kt)
        Am[tl][kt] = ldA_f32(mr + (size_t)(nbase + tl * 16) * CIN + kt * 32, CIN, lane);

    // --- G = sigmoid(m@Wg+bg), P = G*o : WgT streamed via LDS in 8 chunks of 64 cols ---
    for (int ch = 0; ch < 8; ++ch) {
      __syncthreads();
      coop_copy_bf16(wbuf, WgT + (size_t)(ch * 64) * CIN, 64 * CIN, tid);
      __syncthreads();
#pragma unroll 2
      for (int jl = 0; jl < 4; ++jl) {
        v8f a0 = {}, a1 = {};
#pragma unroll
        for (int kt = 0; kt < 8; ++kt) {
          v16bf Bf = ldB_T_bf16(wbuf + (size_t)(jl * 16) * CIN + kt * 32, CIN, lane);
          a0 = wmma_bf16(Am[0][kt], Bf, a0);
          a1 = wmma_bf16(Am[1][kt], Bf, a1);
        }
        int j = ch * 64 + jl * 16 + colL;
        float oj = olds[j], bgj = bg[j];
#pragma unroll
        for (int i = 0; i < 8; ++i) {
          float g0 = 1.f / (1.f + __expf(-(a0[i] + bgj)));
          float g1 = 1.f / (1.f + __expf(-(a1[i] + bgj)));
          pst[(rb + i) * HC + j]        = f2bf(g0 * oj);
          pst[(16 + rb + i) * HC + j]   = f2bf(g1 * oj);
        }
      }
    }

    // --- out = P@Wo + bo : WoT streamed via LDS in 8 chunks of 32 out-cols ---
    float* op = out + ((size_t)r * N_ + nbase) * CIN;
    for (int ch = 0; ch < 8; ++ch) {
      __syncthreads();
      coop_copy_bf16(wbuf, WoT + (size_t)(ch * 32) * HC, 32 * HC, tid);
      __syncthreads();
#pragma unroll
      for (int jl = 0; jl < 2; ++jl) {
        v8f a0 = {}, a1 = {};
#pragma unroll
        for (int kt = 0; kt < 16; ++kt) {
          v16bf Bf = ldB_T_bf16(wbuf + (size_t)(jl * 16) * HC + kt * 32, HC, lane);
          a0 = wmma_bf16(ldA_bf16(pst + kt * 32, HC, lane), Bf, a0);
          a1 = wmma_bf16(ldA_bf16(pst + 16 * HC + kt * 32, HC, lane), Bf, a1);
        }
        int j = ch * 32 + jl * 16 + colL;
        float boj = bo[j];
#pragma unroll
        for (int i = 0; i < 8; ++i) {
          op[(size_t)(rb + i) * CIN + j]        = a0[i] + boj;
          op[(size_t)(16 + rb + i) * CIN + j]   = a1[i] + boj;
        }
      }
    }
  }
}

// =====================  host launcher  =====================
extern "C" void kernel_launch(void* const* d_in, const int* in_sizes, int n_in,
                              void* d_out, int out_size, void* d_ws, size_t ws_size,
                              hipStream_t stream) {
  const float* m    = (const float*)d_in[0];
  const float* mask = (const float*)d_in[1];
  const float* Wq   = (const float*)d_in[2];
  const float* Wk   = (const float*)d_in[3];
  const float* Wv   = (const float*)d_in[4];
  const float* Wg   = (const float*)d_in[5];
  const float* bg   = (const float*)d_in[6];
  const float* Wo   = (const float*)d_in[7];
  const float* bo   = (const float*)d_in[8];
  float* out = (float*)d_out;

  char* ws = (char*)d_ws;
  __bf16* WgT = (__bf16*)(ws);                     // 512x256 bf16 = 262144 B
  __bf16* WoT = (__bf16*)(ws + 262144);            // 256x512 bf16 = 262144 B
  __bf16* WvT = (__bf16*)(ws + 524288);            // 64x256 bf16  =  32768 B

  // weight conversions (transposed, B-fragment friendly)
  wconv_kernel<<<(CIN * HC + 255) / 256, 256, 0, stream>>>(Wg, WgT, CIN, HC);
  wconv_kernel<<<(HC * CIN + 255) / 256, 256, 0, stream>>>(Wo, WoT, HC, CIN);
  wconv_kernel<<<(CIN * CH + 255) / 256, 256, 0, stream>>>(Wv, WvT, CIN, CH);

  // fully fused global attention: one workgroup (8 wave32) per residue row
  hipFuncSetAttribute((const void*)attn_row_kernel,
                      hipFuncAttributeMaxDynamicSharedMemorySize, SMEM_BYTES);
  attn_row_kernel<<<R_, 256, SMEM_BYTES, stream>>>(m, mask, Wq, Wk, WgT, WoT, WvT,
                                                   bg, bo, out);
}
